// AttnMessageAggregator_72052371357815
// MI455X (gfx1250) — matrix operand order repaired and checked by reading, not compiled, not run on hardware
//
#include <hip/hip_runtime.h>
#include <math.h>

// ---------------- problem constants ----------------
static constexpr int Bc  = 32768;
static constexpr int Nc  = 20;
static constexpr int Dc  = 172;
static constexpr int Tc  = 100;
static constexpr int Ec  = 272;      // D + T
static constexpr int HDc = 136;      // E / 2 heads
static constexpr int KP  = 288;      // 272 padded to 9*32  (K for E-GEMMs)
static constexpr int K1  = 448;      // 444 padded to 14*32 (K for fc1)
static constexpr int K2  = 192;      // 172 padded to 6*32  (K for fc2)
static constexpr int H1W = 448;      // fc1 input row stride (272 attn_out + 172 source + 4 pad)

typedef __bf16 bf16_t;
typedef __attribute__((ext_vector_type(8)))  __bf16 v8bf;
typedef __attribute__((ext_vector_type(16))) __bf16 v16bf;
typedef __attribute__((ext_vector_type(8)))  float  v8f;

// ---------------- workspace layout (bytes) ----------------
static constexpr size_t al256(size_t x) { return (x + 255) & ~size_t(255); }
static constexpr size_t OFF_WKV  = 0;                                          // 544*288 bf16
static constexpr size_t OFF_WQ   = al256(OFF_WKV  + size_t(2*Ec)*KP*2);        // 272*288 bf16
static constexpr size_t OFF_WOUT = al256(OFF_WQ   + size_t(Ec)*KP*2);          // 272*288 bf16
static constexpr size_t OFF_WFC1 = al256(OFF_WOUT + size_t(Ec)*KP*2);          // 172*448 bf16
static constexpr size_t OFF_WFC2 = al256(OFF_WFC1 + size_t(Dc)*K1*2);          // 172*192 bf16
static constexpr size_t OFF_MASK = al256(OFF_WFC2 + size_t(Dc)*K2*2);          // B*N u8
static constexpr size_t OFF_INV  = al256(OFF_MASK + size_t(Bc)*Nc);            // B u8
static constexpr size_t OFF_QMAT = al256(OFF_INV  + size_t(Bc));               // B*288 bf16
static constexpr size_t OFF_QP   = al256(OFF_QMAT + size_t(Bc)*KP*2);          // B*272 bf16
static constexpr size_t OFF_CTX  = al256(OFF_QP   + size_t(Bc)*Ec*2);          // B*288 bf16
static constexpr size_t OFF_H1   = al256(OFF_CTX  + size_t(Bc)*KP*2);          // B*448 bf16
static constexpr size_t OFF_HID  = al256(OFF_H1   + size_t(Bc)*H1W*2);         // B*192 bf16
static constexpr size_t OFF_MSGS = al256(OFF_HID  + size_t(Bc)*K2*2);          // B*N*288 bf16
static constexpr size_t OFF_KV   = al256(OFF_MSGS + size_t(Bc)*Nc*KP*2);       // B*N*544 bf16

// =====================================================================
// Kernel 1: weights fp32 -> zero-padded bf16
// =====================================================================
__global__ __launch_bounds__(256)
void prep_weights(const float* __restrict__ in_proj_w,
                  const float* __restrict__ out_proj_w,
                  const float* __restrict__ fc1_w,
                  const float* __restrict__ fc2_w,
                  bf16_t* __restrict__ wkv,  bf16_t* __restrict__ wq,
                  bf16_t* __restrict__ wout, bf16_t* __restrict__ wfc1,
                  bf16_t* __restrict__ wfc2)
{
    int idx = blockIdx.x * blockDim.x + threadIdx.x;
    const int S0 = 2*Ec*KP, S1 = Ec*KP, S2 = Ec*KP, S3 = Dc*K1, S4 = Dc*K2;
    if (idx < S0) {                       // wk rows 0..271 then wv rows 272..543
        int j = idx / KP, k = idx % KP;
        float v = (k < Ec) ? in_proj_w[(size_t)(Ec + j) * Ec + k] : 0.f;
        wkv[idx] = (bf16_t)v; return;
    }
    idx -= S0;
    if (idx < S1) { int j = idx / KP, k = idx % KP;
        wq[idx]   = (bf16_t)((k < Ec) ? in_proj_w[(size_t)j*Ec + k] : 0.f); return; }
    idx -= S1;
    if (idx < S2) { int j = idx / KP, k = idx % KP;
        wout[idx] = (bf16_t)((k < Ec) ? out_proj_w[(size_t)j*Ec + k] : 0.f); return; }
    idx -= S2;
    if (idx < S3) { int j = idx / K1, k = idx % K1;
        wfc1[idx] = (bf16_t)((k < Ec+Dc) ? fc1_w[(size_t)j*(Ec+Dc) + k] : 0.f); return; }
    idx -= S3;
    if (idx < S4) { int j = idx / K2, k = idx % K2;
        wfc2[idx] = (bf16_t)((k < Dc) ? fc2_w[(size_t)j*Dc + k] : 0.f); return; }
}

// =====================================================================
// Kernel 2: assemble msgs_un (bf16, padded), Q rows, source half of fc1
//           input, adjusted mask and invalid flags. One wave per row;
//           rows b*(N+1)..: r<N -> msg row, r==N -> per-b work.
// =====================================================================
__global__ __launch_bounds__(256)
void assemble(const float* __restrict__ msg,   const float* __restrict__ timeb,
              const float* __restrict__ time_w, const float* __restrict__ time_b,
              bf16_t* __restrict__ msgs, bf16_t* __restrict__ qmat,
              bf16_t* __restrict__ h1in, unsigned char* __restrict__ maskb,
              unsigned char* __restrict__ invb)
{
    int tid  = blockIdx.x * blockDim.x + threadIdx.x;
    int wid  = tid >> 5, lane = tid & 31;
    int b = wid / (Nc + 1);
    int r = wid % (Nc + 1);
    if (b >= Bc) return;
    float t_src = timeb[(size_t)b*Nc + (Nc-1)];

    if (r < Nc) {
        size_t row = (size_t)b*Nc + r;
        float delta = timeb[row] - t_src;
        const float* mrow = msg + row * Dc;
        bf16_t* o = msgs + row * KP;
        for (int c = lane; c < KP; c += 32) {
            float v;
            if      (c < Dc) v = mrow[c];
            else if (c < Ec) v = __cosf(delta * time_w[c-Dc] + time_b[c-Dc]);
            else             v = 0.f;
            o[c] = (bf16_t)v;
        }
    } else {
        const float* src = msg + ((size_t)b*Nc + (Nc-1)) * Dc;
        // Q row: [source | cos(time_b)] padded
        bf16_t* q = qmat + (size_t)b * KP;
        for (int c = lane; c < KP; c += 32) {
            float v;
            if      (c < Dc) v = src[c];
            else if (c < Ec) v = __cosf(time_b[c-Dc]);   // delta == 0 for source row
            else             v = 0.f;
            q[c] = (bf16_t)v;
        }
        // fc1 input, columns E..447: source then zero pad
        bf16_t* h1 = h1in + (size_t)b * H1W + Ec;
        for (int c = lane; c < (H1W - Ec); c += 32)
            h1[c] = (c < Dc) ? (bf16_t)src[c] : (bf16_t)0.f;
        // mask / invalid (wave-wide AND over n<20)
        unsigned mk = 1u;
        if (lane < Nc) mk = (timeb[(size_t)b*Nc + lane] < 0.f) ? 1u : 0u;
        unsigned allm = mk;
#pragma unroll
        for (int o2 = 16; o2 > 0; o2 >>= 1)
            allm &= (unsigned)__shfl_xor((int)allm, o2, 32);
        bool inval = (allm != 0u);
        if (lane < Nc) {
            unsigned char mo = (unsigned char)mk;
            if (lane == 0 && inval) mo = 0;              // un-mask n=0 when all masked
            maskb[(size_t)b*Nc + lane] = mo;
        }
        if (lane == 0) invb[b] = inval ? 1 : 0;
    }
}

// =====================================================================
// Generic bf16 WMMA GEMM:  out(M,Nout) = A(M,K) * W(Nout,K)^T + bias
// One wave computes a 16x64 tile (4 x v_wmma_f32_16x16x32_bf16 per K-step,
// A fragment reused 4x). Software-pipelined: fragment loads for k+1 are
// issued before the k-step WMMAs so the matrix pipe overlaps VMEM latency
// (waits become loadcnt<=N instead of 0). A and W fragments are
// contiguous b128 loads. N-edge: weight-row loads clamped (EXEC stays
// all-ones for WMMA), stores predicated.
// =====================================================================
enum { F_OUT_F32 = 1, F_OUT_BF16 = 2, F_RELU = 4, F_ZINV = 8 };

template<int FLAGS>
__global__ __launch_bounds__(256)
void gemm_bf16_wmma(const bf16_t* __restrict__ A, int lda,
                    const bf16_t* __restrict__ W, int ldw,
                    const float* __restrict__ bias,
                    const unsigned char* __restrict__ inval,
                    void* __restrict__ outp, int ldo,
                    int M, int Nout, int ksteps, int ntiles)
{
    constexpr int NT = 4;
    int tid  = blockIdx.x * blockDim.x + threadIdx.x;
    int wid  = tid >> 5;
    int lane = tid & 31;
    int mt = wid / ntiles;
    int nt = wid % ntiles;
    if (mt * 16 >= M) return;            // wave-uniform exit

    int l15 = lane & 15;
    int hs  = lane >> 4;                 // half-select: lanes 16..31

    const bf16_t* aptr = A + (size_t)(mt*16 + l15) * lda;
    const bf16_t* wptr[NT];
#pragma unroll
    for (int s = 0; s < NT; ++s) {
        int col = nt*(16*NT) + s*16 + l15;
        int rr  = (col < Nout) ? col : (Nout - 1);   // clamp, keep loads valid
        wptr[s] = W + (size_t)rr * ldw;
    }

    v8f acc[NT] = {};

    // -------- software-pipelined K loop (register double buffer) --------
    // A fragment: lanes<16 hold K[0..7],[16..23]; lanes>=16 K[8..15],[24..31]
    // B fragment: lanes<16 hold K[0..7],[16..23]; lanes>=16 K[8..15],[24..31] of column n
    v8bf a0 = *(const v8bf*)(aptr + hs*8);
    v8bf a1 = *(const v8bf*)(aptr + hs*8 + 16);
    v8bf b0[NT], b1[NT];
#pragma unroll
    for (int s = 0; s < NT; ++s) {
        b0[s] = *(const v8bf*)(wptr[s] + hs*8);
        b1[s] = *(const v8bf*)(wptr[s] + hs*8 + 16);
    }

    auto do_wmma = [&](v8bf A0, v8bf A1, v8bf (&B0)[NT], v8bf (&B1)[NT]) {
        v16bf af = __builtin_shufflevector(A0, A1,
                     0,1,2,3,4,5,6,7,8,9,10,11,12,13,14,15);
#pragma unroll
        for (int s = 0; s < NT; ++s) {
            v16bf bfv = __builtin_shufflevector(B0[s], B1[s],
                          0,1,2,3,4,5,6,7,8,9,10,11,12,13,14,15);
            acc[s] = __builtin_amdgcn_wmma_f32_16x16x32_bf16(
                         false, af, false, bfv, (short)0, acc[s], false, false);
        }
    };

    for (int kk = 0; kk + 1 < ksteps; ++kk) {
        int kn = (kk + 1) * 32;
        __builtin_prefetch(aptr + kn + 32, 0, 3);    // one k-step ahead of loads
        // issue next-iteration fragment loads first ...
        v8bf na0 = *(const v8bf*)(aptr + kn + hs*8);
        v8bf na1 = *(const v8bf*)(aptr + kn + hs*8 + 16);
        v8bf nb0[NT], nb1[NT];
#pragma unroll
        for (int s = 0; s < NT; ++s) {
            nb0[s] = *(const v8bf*)(wptr[s] + kn + hs*8);
            nb1[s] = *(const v8bf*)(wptr[s] + kn + hs*8 + 16);
        }
        // ... then compute on the current buffers while they are in flight
        do_wmma(a0, a1, b0, b1);
        a0 = na0; a1 = na1;
#pragma unroll
        for (int s = 0; s < NT; ++s) { b0[s] = nb0[s]; b1[s] = nb1[s]; }
    }
    do_wmma(a0, a1, b0, b1);             // drain last k-step

    // -------- epilogue --------
#pragma unroll
    for (int s = 0; s < NT; ++s) {
        int col = nt*(16*NT) + s*16 + l15;
        if (col >= Nout) continue;
        float bv = bias ? bias[col] : 0.f;
#pragma unroll
        for (int r = 0; r < 8; ++r) {
            int m = mt*16 + hs*8 + r;                // C layout: VGPR r -> M = r (+8)
            float v = acc[s][r] + bv;
            if (FLAGS & F_RELU) v = fmaxf(v, 0.f);
            if (FLAGS & F_ZINV) { if (inval[m]) v = 0.f; }
            if (FLAGS & F_OUT_F32)
                ((float*)outp)[(size_t)m * ldo + col] = v;
            else
                ((bf16_t*)outp)[(size_t)m * ldo + col] = (bf16_t)v;
        }
    }
}

// =====================================================================
// Kernel: attention. One wave per (b, head). Lane-parallel over HD=136,
// butterfly __shfl_xor reductions for the q.k dots, redundant softmax
// over N=20 per lane, then ctx = attn @ v written as bf16 into CTX.
// =====================================================================
__global__ __launch_bounds__(256)
void attn_kernel(const bf16_t* __restrict__ qp, const bf16_t* __restrict__ kv,
                 const unsigned char* __restrict__ maskb, bf16_t* __restrict__ ctx)
{
    int tid  = blockIdx.x * blockDim.x + threadIdx.x;
    int wid  = tid >> 5, lane = tid & 31;
    int b = wid >> 1, h = wid & 1;
    if (b >= Bc) return;

    const bf16_t* q = qp + (size_t)b * Ec + h * HDc;
    float qv[5];
#pragma unroll
    for (int j = 0; j < 5; ++j) {
        int d = lane + 32*j;
        qv[j] = (d < HDc) ? (float)q[d] : 0.f;
    }

    const float scale = 0.08574929257125441f;   // 1/sqrt(136)
    float sc[Nc];
    for (int n = 0; n < Nc; ++n) {
        const bf16_t* kr = kv + ((size_t)b*Nc + n) * (2*Ec) + h * HDc;
        float p = 0.f;
#pragma unroll
        for (int j = 0; j < 5; ++j) {
            int d = lane + 32*j;
            if (d < HDc) p += qv[j] * (float)kr[d];
        }
#pragma unroll
        for (int o = 16; o > 0; o >>= 1) p += __shfl_xor(p, o, 32);
        p *= scale;
        if (maskb[(size_t)b*Nc + n]) p = -1e9f;
        sc[n] = p;
    }
    float mx = sc[0];
#pragma unroll
    for (int n = 1; n < Nc; ++n) mx = fmaxf(mx, sc[n]);
    float den = 0.f;
#pragma unroll
    for (int n = 0; n < Nc; ++n) { sc[n] = __expf(sc[n] - mx); den += sc[n]; }
    float rden = 1.f / den;

    float cx[5] = {0.f, 0.f, 0.f, 0.f, 0.f};
    for (int n = 0; n < Nc; ++n) {
        const bf16_t* vr = kv + ((size_t)b*Nc + n) * (2*Ec) + Ec + h * HDc;
        float w = sc[n] * rden;
#pragma unroll
        for (int j = 0; j < 5; ++j) {
            int d = lane + 32*j;
            if (d < HDc) cx[j] += w * (float)vr[d];
        }
    }
#pragma unroll
    for (int j = 0; j < 5; ++j) {
        int d = lane + 32*j;
        if (d < HDc) ctx[(size_t)b*KP + h*HDc + d] = (bf16_t)cx[j];
    }
}

// =====================================================================
// launch
// =====================================================================
extern "C" void kernel_launch(void* const* d_in, const int* in_sizes, int n_in,
                              void* d_out, int out_size, void* d_ws, size_t ws_size,
                              hipStream_t stream)
{
    (void)in_sizes; (void)n_in; (void)out_size; (void)ws_size;
    const float* message_batch = (const float*)d_in[0];
    const float* time_batch    = (const float*)d_in[1];
    /* d_in[2] 'memory' is unused by the reference */
    const float* time_w     = (const float*)d_in[3];
    const float* time_b     = (const float*)d_in[4];
    const float* in_proj_w  = (const float*)d_in[5];
    const float* in_proj_b  = (const float*)d_in[6];
    const float* out_proj_w = (const float*)d_in[7];
    const float* out_proj_b = (const float*)d_in[8];
    const float* fc1_w      = (const float*)d_in[9];
    const float* fc1_b      = (const float*)d_in[10];
    const float* fc2_w      = (const float*)d_in[11];
    const float* fc2_b      = (const float*)d_in[12];

    char* ws = (char*)d_ws;
    bf16_t* WKV  = (bf16_t*)(ws + OFF_WKV);
    bf16_t* WQ   = (bf16_t*)(ws + OFF_WQ);
    bf16_t* WOUT = (bf16_t*)(ws + OFF_WOUT);
    bf16_t* WFC1 = (bf16_t*)(ws + OFF_WFC1);
    bf16_t* WFC2 = (bf16_t*)(ws + OFF_WFC2);
    unsigned char* MASK = (unsigned char*)(ws + OFF_MASK);
    unsigned char* INV  = (unsigned char*)(ws + OFF_INV);
    bf16_t* QMAT = (bf16_t*)(ws + OFF_QMAT);
    bf16_t* QP   = (bf16_t*)(ws + OFF_QP);
    bf16_t* CTX  = (bf16_t*)(ws + OFF_CTX);
    bf16_t* H1IN = (bf16_t*)(ws + OFF_H1);
    bf16_t* HID  = (bf16_t*)(ws + OFF_HID);
    bf16_t* MSGS = (bf16_t*)(ws + OFF_MSGS);
    bf16_t* KV   = (bf16_t*)(ws + OFF_KV);

    // 1. weights -> bf16 (zero K-padded)
    {
        int total = 2*Ec*KP + Ec*KP + Ec*KP + Dc*K1 + Dc*K2;
        prep_weights<<<(total + 255)/256, 256, 0, stream>>>(
            in_proj_w, out_proj_w, fc1_w, fc2_w, WKV, WQ, WOUT, WFC1, WFC2);
    }
    // 2. assemble activations / masks
    {
        long waves  = (long)Bc * (Nc + 1);
        long blocks = (waves + 7) / 8;
        assemble<<<blocks, 256, 0, stream>>>(
            message_batch, time_batch, time_w, time_b,
            MSGS, QMAT, H1IN, MASK, INV);
    }
    // 3. KV projection: (B*N,288) @ (288,544)   [bk|bv = in_proj_b[E..3E)]
    {
        int M = Bc*Nc, Nout = 2*Ec, ntiles = (Nout + 63)/64;
        long waves = (long)((M + 15)/16) * ntiles;
        gemm_bf16_wmma<F_OUT_BF16><<<(waves + 7)/8, 256, 0, stream>>>(
            MSGS, KP, WKV, KP, in_proj_b + Ec, nullptr,
            KV, 2*Ec, M, Nout, KP/32, ntiles);
    }
    // 4. Q projection: (B,288) @ (288,272)
    {
        int Nout = Ec, ntiles = (Nout + 63)/64;
        long waves = (long)(Bc/16) * ntiles;
        gemm_bf16_wmma<F_OUT_BF16><<<(waves + 7)/8, 256, 0, stream>>>(
            QMAT, KP, WQ, KP, in_proj_b, nullptr,
            QP, Ec, Bc, Nout, KP/32, ntiles);
    }
    // 5. attention
    {
        long waves = (long)Bc * 2;
        attn_kernel<<<(waves + 7)/8, 256, 0, stream>>>(QP, KV, MASK, CTX);
    }
    // 6. out_proj: (B,288) @ (288,272), +bias, zero-if-invalid, -> fc1 input cols [0,272)
    {
        int Nout = Ec, ntiles = (Nout + 63)/64;
        long waves = (long)(Bc/16) * ntiles;
        gemm_bf16_wmma<F_OUT_BF16 | F_ZINV><<<(waves + 7)/8, 256, 0, stream>>>(
            CTX, KP, WOUT, KP, out_proj_b, INV,
            H1IN, H1W, Bc, Nout, KP/32, ntiles);
    }
    // 7. fc1: (B,448) @ (448,172), +bias, relu
    {
        int Nout = Dc, ntiles = (Nout + 63)/64;
        long waves = (long)(Bc/16) * ntiles;
        gemm_bf16_wmma<F_OUT_BF16 | F_RELU><<<(waves + 7)/8, 256, 0, stream>>>(
            H1IN, K1, WFC1, K1, fc1_b, nullptr,
            HID, K2, Bc, Nout, K1/32, ntiles);
    }
    // 8. fc2: (B,192) @ (192,172), +bias -> d_out f32
    {
        int Nout = Dc, ntiles = (Nout + 63)/64;
        long waves = (long)(Bc/16) * ntiles;
        gemm_bf16_wmma<F_OUT_F32><<<(waves + 7)/8, 256, 0, stream>>>(
            HID, K2, WFC2, K2, fc2_b, nullptr,
            d_out, Dc, Bc, Nout, K2/32, ntiles);
    }
}
